// Point_Transformer_Last_19507741458796
// MI455X (gfx1250) — matrix operand order repaired and checked by reading, not compiled, not run on hardware
//
#include <hip/hip_runtime.h>
#include <stdint.h>

#define B_ 16
#define C_ 768
#define N_ 1024
#define L_ 16
#define EPS_ 0.25f
#define DT_ 0.03f

typedef __bf16 bf16_t;
typedef __attribute__((ext_vector_type(16))) bf16_t v16bf;
typedef __attribute__((ext_vector_type(8)))  float  v8f;

union V16BF { v16bf v; uint4 q[2]; };

__device__ __forceinline__ unsigned int pack_bf2(float a, float b) {
  unsigned int ua = __float_as_uint(a);
  unsigned int ub = __float_as_uint(b);
  ua = (ua + 0x7FFFu + ((ua >> 16) & 1u)) >> 16;   // RNE f32 -> bf16
  ub = (ub + 0x7FFFu + ((ub >> 16) & 1u)) >> 16;
  return (ua & 0xFFFFu) | (ub << 16);
}

__device__ __forceinline__ float dot4(float4 a, float4 b) {
  return a.x * b.x + a.y * b.y + a.z * b.z + a.w * b.w;
}

// ---------------- Kernel 1: f32 x (B,1,C,N) -> bf16 xbf (B,C,N) ----------------
__global__ __launch_bounds__(256) void cvt_kernel(const float* __restrict__ x,
                                                  unsigned int* __restrict__ o) {
  const int i = blockIdx.x * 256 + threadIdx.x;     // exact: B*C*N/2 threads
  const float2 v = ((const float2*)x)[i];
  o[i] = pack_bf2(v.x, v.y);
}

// ---------------- Kernel 2: z[b,n,l] = xo[b,:,n] . proj_w[l,:] + pb[l]; sq[b,n] ----------------
__global__ __launch_bounds__(256) void z_kernel(const float* __restrict__ xo,
                                                const float* __restrict__ pw,
                                                const float* __restrict__ pb,
                                                float* __restrict__ z,
                                                float* __restrict__ sq) {
  const int l  = threadIdx.x;                 // 0..15
  const int ny = threadIdx.y;                 // 0..15
  const int n  = blockIdx.x * 16 + ny;
  const int b  = blockIdx.y;
  const float* xp = xo + (size_t)b * C_ * N_ + n;
  const float* wp = pw + (size_t)l * C_;
  float acc = pb[l];
#pragma unroll 4
  for (int c = 0; c < C_; ++c) acc += xp[(size_t)c * N_] * wp[c];
  z[((size_t)b * N_ + n) * L_ + l] = acc;
  __shared__ float s[16][17];
  s[ny][l] = acc * acc;
  __syncthreads();
  if (l == 0) {
    float t = 0.f;
#pragma unroll
    for (int k = 0; k < 16; ++k) t += s[ny][k];
    sq[(size_t)b * N_ + n] = t;
  }
}

// ---------------- Kernel 3: pi = sigmoid( relu(z@w1^T + b1) @ w2^T + b2 ) ----------------
__global__ __launch_bounds__(256) void pi_kernel(const float* __restrict__ z,
                                                 const float* __restrict__ w1,
                                                 const float* __restrict__ b1,
                                                 const float* __restrict__ w2,
                                                 const float* __restrict__ b2,
                                                 float* __restrict__ pi) {
  const int gid = blockIdx.x * 256 + threadIdx.x;   // b*N + n
  const float4* z4 = (const float4*)(z + (size_t)gid * L_);
  const float4 z0 = z4[0], z1 = z4[1], z2 = z4[2], z3 = z4[3];
  float acc = 0.f;
#pragma unroll 2
  for (int c = 0; c < C_; ++c) {
    const float4* w4 = (const float4*)(w1 + (size_t)c * L_);
    float h = b1[c] + dot4(z0, w4[0]) + dot4(z1, w4[1]) + dot4(z2, w4[2]) + dot4(z3, w4[3]);
    h = fmaxf(h, 0.f);
    acc += h * w2[c];
  }
  acc += b2[0];
  pi[gid] = 1.0f / (1.0f + expf(-acc));
}

// ---------------- Kernel 4: q[b,n] = sum_j exp(-sqdist); col = pi/q ----------------
__global__ __launch_bounds__(256) void qcol_kernel(const float* __restrict__ z,
                                                   const float* __restrict__ sq,
                                                   const float* __restrict__ pi,
                                                   float* __restrict__ col) {
  const int t = threadIdx.x;
  const int lane = t & 31;
  const int wave = t >> 5;
  const int gid = blockIdx.x * 8 + wave;            // b*N + n
  const int b = gid >> 10;
  const int n = gid & (N_ - 1);
  const float* zb  = z  + (size_t)b * N_ * L_;
  const float* sqb = sq + (size_t)b * N_;
  const float4* zn4 = (const float4*)(zb + (size_t)n * L_);
  const float4 z0 = zn4[0], z1 = zn4[1], z2 = zn4[2], z3 = zn4[3];
  const float sqn = sqb[n];
  float acc = 0.f;
  for (int j = lane; j < N_; j += 32) {
    const float4* zj4 = (const float4*)(zb + (size_t)j * L_);
    const float d = dot4(z0, zj4[0]) + dot4(z1, zj4[1]) + dot4(z2, zj4[2]) + dot4(z3, zj4[3]);
    const float g = sqn + sqb[j] - 2.f * d;
    acc += expf(-fmaxf(g, 0.f));
  }
#pragma unroll
  for (int off = 16; off >= 1; off >>= 1) acc += __shfl_xor(acc, off, 32);
  if (lane == 0) col[gid] = pi[gid] / acc;
}

// ---------------- Kernel 5: fused K-tile build + bf16 WMMA GEMM + diffusion ----------------
// grid (N/16, B), block 256 (8 waves). Workgroup owns a 16-row tile; each wave owns
// 6 output column tiles (6*16 = 96 cols, 8 waves -> 768). Double-buffered A tile in LDS,
// all 12 B-tile loads issued before the barrier so L2 latency overlaps barrier + LDS read.
__global__ __launch_bounds__(256) void diffuse_kernel(const float* __restrict__ x,
                                                      const unsigned short* __restrict__ xbf,
                                                      const float* __restrict__ z,
                                                      const float* __restrict__ sq,
                                                      const float* __restrict__ col,
                                                      float* __restrict__ out) {
  __shared__ float zi[256];
  __shared__ float sqi[16];
  __shared__ __align__(16) unsigned int Abuf[2][256];  // double-buffered 16x32 bf16 A tile
  __shared__ float dpart[256];
  __shared__ __align__(16) float dinv[16];

  const int t = threadIdx.x;
  const int lane = t & 31;
  const int wave = t >> 5;
  const int itile = blockIdx.x;
  const int b = blockIdx.y;
  const int i0 = itile * 16;

  const float* zb  = z   + (size_t)b * N_ * L_;
  const float* sqb = sq  + (size_t)b * N_;
  const float* clb = col + (size_t)b * N_;
  const unsigned short* xb = xbf + (size_t)b * C_ * N_;

  zi[t] = zb[(size_t)i0 * L_ + t];
  if (t < 16) sqi[t] = sqb[i0 + t];
  __syncthreads();

  const int r  = t & 15;          // row of S tile this thread computes
  const int jc = t >> 4;          // which j-pair (K = 2*jc, 2*jc+1)
  const float4* zr4 = (const float4*)&zi[r * 16];
  const float4 zr0 = zr4[0], zr1 = zr4[1], zr2 = zr4[2], zr3 = zr4[3];
  const float sqr = sqi[r];

  // A-operand LDS slot (contiguous-K layout matching B: lane r holds K 0..15;
  // lane r+16 holds K 16..31; two bf16 halves per dword)
  const int K = 2 * jc;
  const int aslot = (r + ((K & 16) ? 16 : 0)) * 8 + ((K & 15) >> 1);

  // B-operand addressing (sequential-K per lane: lanes 0-15 -> K 0..15, 16-31 -> K 16..31)
  const int cLane = lane & 15;
  const int kOff  = (lane < 16) ? 0 : 16;

  // Per-wave B base pointers for its 6 column tiles, advanced by 32 elements per j-step.
  const unsigned short* bptr[6];
#pragma unroll
  for (int k = 0; k < 6; ++k)
    bptr[k] = xb + (size_t)((wave * 6 + k) * 16 + cLane) * N_ + kOff;

  const v8f vzero = {0.f, 0.f, 0.f, 0.f, 0.f, 0.f, 0.f, 0.f};
  v8f acc[6];
#pragma unroll
  for (int k = 0; k < 6; ++k) acc[k] = vzero;
  float accD = 0.f;

  for (int j0 = 0; j0 < N_; j0 += 32) {
    const int p = (j0 >> 5) & 1;
    // ---- compute phase: S[r][K], S[r][K+1] in f32, scale by col, pack bf16 ----
    const int j = j0 + K;
    const float4* zj4 = (const float4*)(zb + (size_t)j * L_);
    const float4 a0 = zj4[0], a1 = zj4[1], a2 = zj4[2], a3 = zj4[3];
    const float4 c0 = zj4[4], c1 = zj4[5], c2 = zj4[6], c3 = zj4[7];
    const float d0 = dot4(zr0, a0) + dot4(zr1, a1) + dot4(zr2, a2) + dot4(zr3, a3);
    const float d1 = dot4(zr0, c0) + dot4(zr1, c1) + dot4(zr2, c2) + dot4(zr3, c3);
    const float g0 = sqr + sqb[j]     - 2.f * d0;
    const float g1 = sqr + sqb[j + 1] - 2.f * d1;
    const float s0 = expf(-fmaxf(g0, 0.f)) * clb[j];
    const float s1 = expf(-fmaxf(g1, 0.f)) * clb[j + 1];
    accD += s0 + s1;
    Abuf[p][aslot] = pack_bf2(s0, s1);

    // ---- issue all 12 B-tile loads before the barrier (independent of LDS) ----
    V16BF Bv[6];
#pragma unroll
    for (int k = 0; k < 6; ++k) {
      const uint4* bp = (const uint4*)(bptr[k] + j0);
      Bv[k].q[0] = bp[0];
      Bv[k].q[1] = bp[1];
    }

    __syncthreads();

    // ---- A from LDS, then the 6-WMMA chain ----
    V16BF A;
    const uint4* Aq = (const uint4*)Abuf[p];
    A.q[0] = Aq[lane * 2];
    A.q[1] = Aq[lane * 2 + 1];
#pragma unroll
    for (int k = 0; k < 6; ++k) {
      acc[k] = __builtin_amdgcn_wmma_f32_16x16x32_bf16(
          false, A.v, false, Bv[k].v, (short)0, acc[k], false, false);
    }
  }

  // ---- reduce d_i over the 16 jc-threads per row ----
  dpart[t] = accD;
  __syncthreads();
  if (t < 16) {
    float s = 0.f;
#pragma unroll
    for (int qq = 0; qq < 16; ++qq) s += dpart[t + 16 * qq];
    dinv[t] = (1.0f / EPS_) / (s + 1e-5f);
  }
  __syncthreads();

  // ---- epilogue: out = xf + DT*((1/eps)*acc/d - xf); C/D layout M = v + 8*(lane>=16) ----
  const int m0 = (lane < 16) ? 0 : 8;
  const float4 dv0 = ((const float4*)dinv)[m0 >> 2];
  const float4 dv1 = ((const float4*)dinv)[(m0 >> 2) + 1];
#pragma unroll
  for (int k = 0; k < 6; ++k) {
    const int cg = (wave * 6 + k) * 16 + cLane;
    const float4* xp = (const float4*)(x   + ((size_t)b * C_ + cg) * N_ + i0 + m0);
    float4*       op = (float4*)      (out + ((size_t)b * C_ + cg) * N_ + i0 + m0);
    const float4 xf0 = xp[0];
    const float4 xf1 = xp[1];
    float4 o0, o1;
    o0.x = xf0.x + DT_ * (acc[k][0] * dv0.x - xf0.x);
    o0.y = xf0.y + DT_ * (acc[k][1] * dv0.y - xf0.y);
    o0.z = xf0.z + DT_ * (acc[k][2] * dv0.z - xf0.z);
    o0.w = xf0.w + DT_ * (acc[k][3] * dv0.w - xf0.w);
    o1.x = xf1.x + DT_ * (acc[k][4] * dv1.x - xf1.x);
    o1.y = xf1.y + DT_ * (acc[k][5] * dv1.y - xf1.y);
    o1.z = xf1.z + DT_ * (acc[k][6] * dv1.z - xf1.z);
    o1.w = xf1.w + DT_ * (acc[k][7] * dv1.w - xf1.w);
    op[0] = o0;
    op[1] = o1;
  }
}

extern "C" void kernel_launch(void* const* d_in, const int* in_sizes, int n_in,
                              void* d_out, int out_size, void* d_ws, size_t ws_size,
                              hipStream_t stream) {
  (void)in_sizes; (void)n_in; (void)out_size; (void)ws_size;
  const float* x  = (const float*)d_in[0];   // (B,1,C,N)
  const float* xo = (const float*)d_in[1];   // (B,C,N)
  const float* pw = (const float*)d_in[2];   // (16,C)
  const float* pb = (const float*)d_in[3];   // (16,)
  const float* w1 = (const float*)d_in[4];   // (C,16)
  const float* b1 = (const float*)d_in[5];   // (C,)
  const float* w2 = (const float*)d_in[6];   // (1,C)
  const float* b2 = (const float*)d_in[7];   // (1,)
  float* out = (float*)d_out;                // (B,C,N)

  char* ws = (char*)d_ws;
  unsigned short* xbf = (unsigned short*)ws;                              // B*C*N bf16
  float* z   = (float*)(ws + (size_t)B_ * C_ * N_ * 2);                   // B*N*16
  float* sq  = (float*)((char*)z  + (size_t)B_ * N_ * L_ * 4);            // B*N
  float* pi  = (float*)((char*)sq + (size_t)B_ * N_ * 4);                 // B*N
  float* col = (float*)((char*)pi + (size_t)B_ * N_ * 4);                 // B*N

  cvt_kernel<<<(B_ * C_ * N_ / 2) / 256, 256, 0, stream>>>(x, (unsigned int*)xbf);
  z_kernel<<<dim3(N_ / 16, B_), dim3(16, 16), 0, stream>>>(xo, pw, pb, z, sq);
  pi_kernel<<<(B_ * N_) / 256, 256, 0, stream>>>(z, w1, b1, w2, b2, pi);
  qcol_kernel<<<(B_ * N_) / 8, 256, 0, stream>>>(z, sq, pi, col);
  diffuse_kernel<<<dim3(N_ / 16, B_), 256, 0, stream>>>(x, xbf, z, sq, col, out);
}
